// Motion_Patch_decoder_var_67946382622873
// MI455X (gfx1250) — matrix operand order repaired and checked by reading, not compiled, not run on hardware
//
#include <hip/hip_runtime.h>
#include <hip/hip_bf16.h>

// ---------------------------------------------------------------------------
// Multi-scale patch decoder for MI455X (gfx1250, wave32).
// All einsums/convs are lowered to a single generic WMMA f16 GEMM
// (v_wmma_f32_16x16x32_f16) with register double-buffering so global_load
// clauses for iteration k+1 overlap the WMMA group of iteration k.
// Workspace requirement: ~207 MB.
// ---------------------------------------------------------------------------

#define BATCH 4
#define LEAK  0.2f

typedef __attribute__((ext_vector_type(16))) _Float16 v16h;
typedef __attribute__((ext_vector_type(8)))  _Float16 v8h;
typedef __attribute__((ext_vector_type(8)))  float    v8f;

static inline int cdiv_h(int a, int b) { return (a + b - 1) / b; }
static inline int round32(int k) { return (k + 31) & ~31; }
static inline int round16(int k) { return (k + 15) & ~15; }

__device__ inline int iclampd(int v, int lo, int hi) { return v < lo ? lo : (v > hi ? hi : v); }

// ---------------- workspace layout (bytes) ----------------
static constexpr size_t OFF_MP     = 0;                         // 65536*800*2
static constexpr size_t OFF_KERN   = 104857600;                 // + 2359296
static constexpr size_t OFF_PACOUT = 107216896;                 // + 8388608
static constexpr size_t OFF_OFFOUT = 115605504;                 // + 13107200
static constexpr size_t OFF_DCNOUT = 128712704;                 // + 4194304
static constexpr size_t OFF_Y      = 132907008;                 // + 4194304
static constexpr size_t OFF_U      = 137101312;                 // + 16777216
static constexpr size_t OFF_STATS  = 153878528;                 // + 4096
static constexpr size_t OFF_WP     = 153882624;                 // + 524288
static constexpr size_t OFF_WO     = 154406912;                 // + 524288
static constexpr size_t OFF_WD     = 154931200;                 // + 524288
static constexpr size_t OFF_WC     = 155455488;                 // + 524288
static constexpr size_t OFF_IMG1   = 155979776;                 // + 16777216
static constexpr size_t OFF_IMG2   = 172756992;                 // + 33554432 -> 206311424 total

// ---------------- weight f32 -> padded f16 [Mpad x Kpad] ----------------
__global__ void wcvt_kernel(const float* __restrict__ w, _Float16* __restrict__ A,
                            int Mreal, int Mpad, int Kreal, int Kpad) {
  int idx = blockIdx.x * 256 + threadIdx.x;
  int tot = Mpad * Kpad;
  if (idx >= tot) return;
  int m = idx / Kpad, k = idx % Kpad;
  float v = (m < Mreal && k < Kreal) ? w[(size_t)m * Kreal + k] : 0.f;
  A[idx] = (_Float16)v;
}

// ---------------- PAC guide kernel: kern[b][pix][9] ----------------
__global__ void pac_kern_kernel(const float* __restrict__ g, float* __restrict__ kern,
                                int C, int H, int W, int total) {
  int idx = blockIdx.x * 256 + threadIdx.x;
  if (idx >= total) return;
  int HW = H * W;
  int b = idx / HW, pix = idx % HW;
  int y = pix / W, x = pix % W;
  float s[9];
#pragma unroll
  for (int t = 0; t < 9; t++) s[t] = 0.f;
  const float* gb = g + (size_t)b * C * HW;
  for (int c = 0; c < C; c++) {
    const float* gc = gb + (size_t)c * HW;
    float ctr = gc[pix];
#pragma unroll
    for (int t = 0; t < 9; t++) {
      int yy = y + t / 3 - 1, xx = x + t % 3 - 1;
      float gv = (yy >= 0 && yy < H && xx >= 0 && xx < W) ? gc[yy * W + xx] : 0.f;
      float d = gv - ctr;
      s[t] += d * d;
    }
  }
  float* ko = kern + ((size_t)b * HW + pix) * 9;
#pragma unroll
  for (int t = 0; t < 9; t++) ko[t] = expf(-0.5f * s[t]);
}

// ---------------- PAC modulated patches -> mp[(b*HW+pix)*Kpad + c*9 + t] ----
__global__ void pac_patch_kernel(const float* __restrict__ xsrc, const float* __restrict__ kern,
                                 _Float16* __restrict__ mp, int C, int H, int W,
                                 int srcH, int srcW, int h0, int w0, int Kpad, int total) {
  int idx = blockIdx.x * 256 + threadIdx.x;
  if (idx >= total) return;
  int c = idx % C;
  int rem = idx / C;
  int HW = H * W;
  int pix = rem % HW, b = rem / HW;
  int y = pix / W, x = pix % W;
  const float* kp = kern + ((size_t)b * HW + pix) * 9;
  const float* xplane = xsrc + (size_t)(b * C + c) * srcH * srcW;
  _Float16* out = mp + ((size_t)b * HW + pix) * Kpad + c * 9;
#pragma unroll
  for (int t = 0; t < 9; t++) {
    int yy = y + t / 3 - 1, xx = x + t % 3 - 1;
    float v = (yy >= 0 && yy < H && xx >= 0 && xx < W)
                  ? xplane[(size_t)(h0 + yy) * srcW + (w0 + xx)] : 0.f;
    out[t] = (_Float16)(v * kp[t]);
  }
}

// ---------------- im2col (3x3, pad 1) from contiguous (B,C,H,W) ----------------
__global__ void im2col_kernel(const float* __restrict__ src, _Float16* __restrict__ mp,
                              int C, int Cth, int H, int W, int Kpad, int total) {
  int idx = blockIdx.x * 256 + threadIdx.x;
  if (idx >= total) return;
  int c = idx % Cth;
  int rem = idx / Cth;
  int HW = H * W;
  int pix = rem % HW, b = rem / HW;
  _Float16* out = mp + ((size_t)b * HW + pix) * Kpad;
  if (c < C) {
    int y = pix / W, x = pix % W;
    const float* plane = src + (size_t)(b * C + c) * HW;
    _Float16* o = out + c * 9;
#pragma unroll
    for (int t = 0; t < 9; t++) {
      int yy = y + t / 3 - 1, xx = x + t % 3 - 1;
      float v = (yy >= 0 && yy < H && xx >= 0 && xx < W) ? plane[yy * W + xx] : 0.f;
      o[t] = (_Float16)v;
    }
  } else {
    for (int k = C * 9; k < Kpad; k++) out[k] = (_Float16)0.f;
  }
}

// ---------------- deformable bilinear sampling -> mp[..][c*kk + k] ----------------
__global__ void dcn_sample_kernel(const float* __restrict__ img, const float* __restrict__ off,
                                  _Float16* __restrict__ mp, int C, int H, int W,
                                  int kd, int kk, int Kpad, int total) {
  int idx = blockIdx.x * 256 + threadIdx.x;
  if (idx >= total) return;
  int k = idx % kk;
  int rem = idx / kk;
  int HW = H * W;
  int pix = rem % HW, b = rem / HW;
  int y = pix / W, x = pix % W;
  float oy = off[((size_t)b * (2 * kk) + 2 * k) * HW + pix];
  float ox = off[((size_t)b * (2 * kk) + 2 * k + 1) * HW + pix];
  int r = kd >> 1;
  float ys = (float)(y + k / kd - r) + oy;
  float xs = (float)(x + k % kd - r) + ox;
  float yf = floorf(ys), xf = floorf(xs);
  float wy = ys - yf, wx = xs - xf;
  int y0 = (int)yf, x0 = (int)xf, y1 = y0 + 1, x1 = x0 + 1;
  float m00 = (y0 >= 0 && y0 < H && x0 >= 0 && x0 < W) ? 1.f : 0.f;
  float m01 = (y0 >= 0 && y0 < H && x1 >= 0 && x1 < W) ? 1.f : 0.f;
  float m10 = (y1 >= 0 && y1 < H && x0 >= 0 && x0 < W) ? 1.f : 0.f;
  float m11 = (y1 >= 0 && y1 < H && x1 >= 0 && x1 < W) ? 1.f : 0.f;
  int cy0 = iclampd(y0, 0, H - 1), cy1 = iclampd(y1, 0, H - 1);
  int cx0 = iclampd(x0, 0, W - 1), cx1 = iclampd(x1, 0, W - 1);
  int o00 = cy0 * W + cx0, o01 = cy0 * W + cx1, o10 = cy1 * W + cx0, o11 = cy1 * W + cx1;
  float w00 = (1.f - wy) * (1.f - wx) * m00, w01 = (1.f - wy) * wx * m01;
  float w10 = wy * (1.f - wx) * m10, w11 = wy * wx * m11;
  const float* base = img + (size_t)b * C * HW;
  _Float16* out = mp + ((size_t)b * HW + pix) * Kpad + k;
  for (int c = 0; c < C; c++) {
    const float* p = base + (size_t)c * HW;
    float v = w00 * p[o00] + w01 * p[o01] + w10 * p[o10] + w11 * p[o11];
    out[(size_t)c * kk] = (_Float16)v;
  }
}

// ---------------- WMMA f16 GEMM: D[b][m][pix] = A[Mpad,Kpad] x Bm[N,Kpad]^T + bias ----
// A: weights (row-major, Kpad per row, zero-padded). Bm: pixel-major patches.
// One wave computes a 16(M) x 64(N) tile; fragments are double-buffered in
// registers so next-iteration loads overlap the current WMMA group.
__global__ __launch_bounds__(256) void gemm_wmma_kernel(
    const _Float16* __restrict__ A, const _Float16* __restrict__ Bm,
    const float* __restrict__ bias, float* __restrict__ D,
    int Mreal, int Mpad, int Kpad, int N, int HW) {
  int wave = (blockIdx.x * 256 + threadIdx.x) >> 5;
  int lane = threadIdx.x & 31;
  int mTiles = Mpad >> 4;
  int nTiles = N >> 6;
  if (wave >= mTiles * nTiles) return;
  int mt = wave % mTiles;
  int nt = wave / mTiles;
  int m0 = mt << 4, n0 = nt << 6;
  int row = lane & 15;
  int hi = lane >> 4;

  const _Float16* Arow = A + (size_t)(m0 + row) * Kpad + (hi ? 16 : 0);
  const _Float16* Brow0 = Bm + (size_t)(n0 + row) * Kpad + (hi ? 16 : 0);
  const _Float16* Brow1 = Brow0 + (size_t)16 * Kpad;
  const _Float16* Brow2 = Brow0 + (size_t)32 * Kpad;
  const _Float16* Brow3 = Brow0 + (size_t)48 * Kpad;

  v8f acc[4];
#pragma unroll
  for (int s = 0; s < 4; s++)
#pragma unroll
    for (int e = 0; e < 8; e++) acc[s][e] = 0.f;

  // prologue: fetch fragments for k = 0
  v8h alo = *(const v8h*)(Arow);
  v8h ahi = *(const v8h*)(Arow + 8);
  v8h blo0 = *(const v8h*)(Brow0), bhi0 = *(const v8h*)(Brow0 + 8);
  v8h blo1 = *(const v8h*)(Brow1), bhi1 = *(const v8h*)(Brow1 + 8);
  v8h blo2 = *(const v8h*)(Brow2), bhi2 = *(const v8h*)(Brow2 + 8);
  v8h blo3 = *(const v8h*)(Brow3), bhi3 = *(const v8h*)(Brow3 + 8);

  for (int k = 0; k < Kpad; k += 32) {
    // snapshot current fragments
    v8h calo = alo, cahi = ahi;
    v8h c0l = blo0, c0h = bhi0, c1l = blo1, c1h = bhi1;
    v8h c2l = blo2, c2h = bhi2, c3l = blo3, c3h = bhi3;
    int kn = k + 32;
    if (kn < Kpad) {  // prefetch next iteration while WMMAs execute
      alo = *(const v8h*)(Arow + kn);
      ahi = *(const v8h*)(Arow + kn + 8);
      blo0 = *(const v8h*)(Brow0 + kn); bhi0 = *(const v8h*)(Brow0 + kn + 8);
      blo1 = *(const v8h*)(Brow1 + kn); bhi1 = *(const v8h*)(Brow1 + kn + 8);
      blo2 = *(const v8h*)(Brow2 + kn); bhi2 = *(const v8h*)(Brow2 + kn + 8);
      blo3 = *(const v8h*)(Brow3 + kn); bhi3 = *(const v8h*)(Brow3 + kn + 8);
    }
    v16h a, b0, b1, b2, b3;
#pragma unroll
    for (int i = 0; i < 8; i++) {
      a[i] = calo[i]; a[i + 8] = cahi[i];
      b0[i] = c0l[i]; b0[i + 8] = c0h[i];
      b1[i] = c1l[i]; b1[i + 8] = c1h[i];
      b2[i] = c2l[i]; b2[i + 8] = c2h[i];
      b3[i] = c3l[i]; b3[i + 8] = c3h[i];
    }
    acc[0] = __builtin_amdgcn_wmma_f32_16x16x32_f16(false, a, false, b0, (short)0, acc[0], false, false);
    acc[1] = __builtin_amdgcn_wmma_f32_16x16x32_f16(false, a, false, b1, (short)0, acc[1], false, false);
    acc[2] = __builtin_amdgcn_wmma_f32_16x16x32_f16(false, a, false, b2, (short)0, acc[2], false, false);
    acc[3] = __builtin_amdgcn_wmma_f32_16x16x32_f16(false, a, false, b3, (short)0, acc[3], false, false);
  }
#pragma unroll
  for (int s = 0; s < 4; s++) {
    int n = n0 + s * 16 + (lane & 15);
    int b = n / HW, pix = n % HW;
#pragma unroll
    for (int r = 0; r < 8; r++) {
      int m = m0 + r + (hi ? 8 : 0);
      if (m < Mreal)
        D[((size_t)b * Mreal + m) * HW + pix] = acc[s][r] + bias[m];
    }
  }
}

// ---------------- 2x bilinear upsample (align-corners style of up_matrix) ----------------
__global__ void up2_kernel(const float* __restrict__ src, float* __restrict__ dst,
                           int Cn, int H, int W, int total) {
  int idx = blockIdx.x * 256 + threadIdx.x;
  if (idx >= total) return;
  int MH = 2 * H, MW = 2 * W;
  int q = idx % MW;
  int rem = idx / MW;
  int p = rem % MH;
  int c = (rem / MH) % Cn;
  int b = rem / (MH * Cn);
  float py = (float)p * (float)(H - 1) / (float)(MH - 1);
  float px = (float)q * (float)(W - 1) / (float)(MW - 1);
  int i0 = (int)floorf(py); if (i0 > H - 2) i0 = H - 2;
  int j0 = (int)floorf(px); if (j0 > W - 2) j0 = W - 2;
  float wy = py - (float)i0, wx = px - (float)j0;
  const float* s = src + (size_t)(b * Cn + c) * H * W;
  float v00 = s[i0 * W + j0], v01 = s[i0 * W + j0 + 1];
  float v10 = s[(i0 + 1) * W + j0], v11 = s[(i0 + 1) * W + j0 + 1];
  float v = (1.f - wy) * ((1.f - wx) * v00 + wx * v01) + wy * ((1.f - wx) * v10 + wx * v11);
  dst[((size_t)(b * Cn + c) * MH + p) * MW + q] = v;
}

// ---------------- instance-norm statistics: one block per (b,c) ----------------
__global__ void inorm_stats_kernel(const float* __restrict__ x, float* __restrict__ stats, int HW) {
  __shared__ float s1[256], s2[256];
  int bc = blockIdx.x, tid = threadIdx.x;
  const float* p = x + (size_t)bc * HW;
  float a = 0.f, b2 = 0.f;
  for (int i = tid; i < HW; i += 256) { float v = p[i]; a += v; b2 += v * v; }
  s1[tid] = a; s2[tid] = b2;
  __syncthreads();
  for (int s = 128; s > 0; s >>= 1) {
    if (tid < s) { s1[tid] += s1[tid + s]; s2[tid] += s2[tid + s]; }
    __syncthreads();
  }
  if (tid == 0) {
    float mu = s1[0] / (float)HW;
    stats[2 * bc] = mu;
    stats[2 * bc + 1] = s2[0] / (float)HW - mu * mu;
  }
}

// ---------------- inorm + leaky + optional residual, written to sub-region ----------------
__global__ void inorm_apply_kernel(const float* __restrict__ x, const float* __restrict__ stats,
                                   const float* __restrict__ gamma, const float* __restrict__ beta,
                                   const float* __restrict__ resid, float* __restrict__ dst,
                                   int Cn, int H, int W, int dstH, int dstW, int h0, int w0,
                                   int total) {
  int idx = blockIdx.x * 256 + threadIdx.x;
  if (idx >= total) return;
  int HW = H * W;
  int pix = idx % HW;
  int rem = idx / HW;
  int c = rem % Cn, b = rem / Cn;
  int y = pix / W, xq = pix % W;
  int bc = b * Cn + c;
  float v = x[(size_t)bc * HW + pix];
  float mu = stats[2 * bc], var = stats[2 * bc + 1];
  float t = (v - mu) * rsqrtf(var + 1e-5f) * gamma[c] + beta[c];
  t = t >= 0.f ? t : LEAK * t;
  size_t di = ((size_t)(b * Cn + c) * dstH + (h0 + y)) * dstW + (w0 + xq);
  dst[di] = resid ? resid[di] + t : t;
}

// ---------------------------------------------------------------------------
// Host-side orchestration
// ---------------------------------------------------------------------------
struct BlockP {
  const float *pac_w, *pac_b, *off_w, *off_b, *dcn_w, *dcn_b, *conv_w, *conv_b, *gamma, *beta;
};

static void gemm_go(const _Float16* A, const _Float16* Bm, const float* bias, float* D,
                    int Mreal, int Mpad, int Kpad, int N, int HW, hipStream_t stream) {
  int waves = (Mpad / 16) * (N / 64);
  gemm_wmma_kernel<<<cdiv_h(waves, 8), 256, 0, stream>>>(A, Bm, bias, D, Mreal, Mpad, Kpad, N, HW);
}

static void run_block(const BlockP& P,
                      const float* xsrc, int srcH, int srcW, int xh0, int xw0,
                      const float* guide,
                      int Cin, int O, int kd, int H, int W,
                      bool do_up,
                      const float* resid, float* dst, int dstH, int dstW, int dh0, int dw0,
                      char* ws, hipStream_t stream) {
  const int HW = H * W, N = BATCH * HW, kk = kd * kd;
  _Float16* mp = (_Float16*)(ws + OFF_MP);
  float* kern = (float*)(ws + OFF_KERN);
  float* pac_out = (float*)(ws + OFF_PACOUT);
  float* off_out = (float*)(ws + OFF_OFFOUT);
  float* dcn_out = (float*)(ws + OFF_DCNOUT);
  float* yb = (float*)(ws + OFF_Y);
  float* ub = (float*)(ws + OFF_U);
  float* stats = (float*)(ws + OFF_STATS);
  _Float16* Wp = (_Float16*)(ws + OFF_WP);
  _Float16* Wo = (_Float16*)(ws + OFF_WO);
  _Float16* Wd = (_Float16*)(ws + OFF_WD);
  _Float16* Wc = (_Float16*)(ws + OFF_WC);

  const int Kp_pac = round32(Cin * 9);
  const int Mp_pac = round16(Cin);
  const int Mp_off = round16(2 * kk);
  const int Kp_dcn = round32(Cin * kk);
  const int Mp_o = round16(O);
  const int Kp_cv = round32(O * 9);

  // weights -> padded f16
  wcvt_kernel<<<cdiv_h(Mp_pac * Kp_pac, 256), 256, 0, stream>>>(P.pac_w, Wp, Cin, Mp_pac, Cin * 9, Kp_pac);
  wcvt_kernel<<<cdiv_h(Mp_off * Kp_pac, 256), 256, 0, stream>>>(P.off_w, Wo, 2 * kk, Mp_off, Cin * 9, Kp_pac);
  wcvt_kernel<<<cdiv_h(Mp_o * Kp_dcn, 256), 256, 0, stream>>>(P.dcn_w, Wd, O, Mp_o, Cin * kk, Kp_dcn);
  wcvt_kernel<<<cdiv_h(Mp_o * Kp_cv, 256), 256, 0, stream>>>(P.conv_w, Wc, O, Mp_o, O * 9, Kp_cv);

  // 1. pac_conv = GEMM over guide-modulated patches
  pac_kern_kernel<<<cdiv_h(BATCH * HW, 256), 256, 0, stream>>>(guide, kern, Cin, H, W, BATCH * HW);
  {
    int tot = BATCH * HW * Cin;
    pac_patch_kernel<<<cdiv_h(tot, 256), 256, 0, stream>>>(xsrc, kern, mp, Cin, H, W,
                                                           srcH, srcW, xh0, xw0, Kp_pac, tot);
  }
  gemm_go(Wp, mp, P.pac_b, pac_out, Cin, Mp_pac, Kp_pac, N, HW, stream);

  // 2. offset conv (3x3) = im2col + GEMM
  {
    int Cth = Cin + (Kp_pac > Cin * 9 ? 1 : 0);
    int tot = BATCH * HW * Cth;
    im2col_kernel<<<cdiv_h(tot, 256), 256, 0, stream>>>(pac_out, mp, Cin, Cth, H, W, Kp_pac, tot);
  }
  gemm_go(Wo, mp, P.off_b, off_out, 2 * kk, Mp_off, Kp_pac, N, HW, stream);

  // 3. deformable sampling + GEMM
  {
    int tot = BATCH * HW * kk;
    dcn_sample_kernel<<<cdiv_h(tot, 256), 256, 0, stream>>>(pac_out, off_out, mp, Cin, H, W,
                                                            kd, kk, Kp_dcn, tot);
  }
  gemm_go(Wd, mp, P.dcn_b, dcn_out, O, Mp_o, Kp_dcn, N, HW, stream);

  // 4. plain 3x3 conv = im2col + GEMM
  {
    int Cth = O + (Kp_cv > O * 9 ? 1 : 0);
    int tot = BATCH * HW * Cth;
    im2col_kernel<<<cdiv_h(tot, 256), 256, 0, stream>>>(dcn_out, mp, O, Cth, H, W, Kp_cv, tot);
  }
  gemm_go(Wc, mp, P.conv_b, yb, O, Mp_o, Kp_cv, N, HW, stream);

  // 5. (up2) + instance norm + leaky (+ residual) into destination region
  const float* nx = yb;
  int nH = H, nW = W;
  if (do_up) {
    int tot = BATCH * O * 4 * HW;
    up2_kernel<<<cdiv_h(tot, 256), 256, 0, stream>>>(yb, ub, O, H, W, tot);
    nx = ub; nH = 2 * H; nW = 2 * W;
  }
  inorm_stats_kernel<<<BATCH * O, 256, 0, stream>>>(nx, stats, nH * nW);
  {
    int tot = BATCH * O * nH * nW;
    inorm_apply_kernel<<<cdiv_h(tot, 256), 256, 0, stream>>>(nx, stats, P.gamma, P.beta,
                                                             resid, dst, O, nH, nW,
                                                             dstH, dstW, dh0, dw0, tot);
  }
}

extern "C" void kernel_launch(void* const* d_in, const int* in_sizes, int n_in,
                              void* d_out, int out_size, void* d_ws, size_t ws_size,
                              hipStream_t stream) {
  const float* patch_1 = (const float*)d_in[0];
  const float* patch_2 = (const float*)d_in[1];
  const float* patch_4 = (const float*)d_in[2];
  const float* g1_0 = (const float*)d_in[3];
  const float* g2[2] = {(const float*)d_in[4], (const float*)d_in[5]};
  const float* g3[4] = {(const float*)d_in[6], (const float*)d_in[7],
                        (const float*)d_in[8], (const float*)d_in[9]};
  // params flattened in dict order: l1_0, l2_0, l2_1, l3_0..l3_3, each
  // {pac_w, pac_b, off_w, off_b, dcn_w, dcn_b, conv_w, conv_b, gamma, beta}
  BlockP blk[7];
  for (int i = 0; i < 7; i++) {
    const float* const* q = (const float* const*)(d_in + 10 + i * 10);
    blk[i].pac_w = q[0]; blk[i].pac_b = q[1];
    blk[i].off_w = q[2]; blk[i].off_b = q[3];
    blk[i].dcn_w = q[4]; blk[i].dcn_b = q[5];
    blk[i].conv_w = q[6]; blk[i].conv_b = q[7];
    blk[i].gamma = q[8]; blk[i].beta = q[9];
  }
  char* ws = (char*)d_ws;
  float* img1 = (float*)(ws + OFF_IMG1);  // (4,64,128,128)
  float* img2 = (float*)(ws + OFF_IMG2);  // (4,32,256,256)
  float* out = (float*)d_out;             // (4,3,256,256)

  // Level 1: patch_1 -> f, img1 = patch_2 + f
  run_block(blk[0], patch_1, 64, 64, 0, 0, g1_0,
            /*Cin*/128, /*O*/64, /*kd*/3, /*H*/64, /*W*/64, /*up*/true,
            patch_2, img1, 128, 128, 0, 0, ws, stream);

  // Level 2: halves of img1 -> f2, img2 = patch_4 + concat(f2)
  for (int j = 0; j < 2; j++)
    run_block(blk[1 + j], img1, 128, 128, j * 64, 0, g2[j],
              64, 32, 3, 64, 128, true,
              patch_4, img2, 256, 256, j * 128, 0, ws, stream);

  // Level 3: quads of img2 -> f3 written directly into d_out
  for (int j = 0; j < 4; j++)
    run_block(blk[3 + j], img2, 256, 256, (j / 2) * 128, (j % 2) * 128, g3[j],
              32, 3, 5, 128, 128, false,
              nullptr, out, 256, 256, (j / 2) * 128, (j % 2) * 128, ws, stream);
}
